// SegmentDecoder_17489106829696
// MI455X (gfx1250) — matrix operand order repaired and checked by reading, not compiled, not run on hardware
//
#include <hip/hip_runtime.h>
#include <cstdint>
#include <cstddef>

#define SEQ   1024
#define EMB   512
#define HID   1024
#define VOCAB 8000
#define G3H   (3 * HID)

typedef __attribute__((ext_vector_type(16))) __bf16 v16bf;
typedef __attribute__((ext_vector_type(8)))  float  v8f;

typedef unsigned int u32x4 __attribute__((ext_vector_type(4)));
typedef int          i32x4 __attribute__((ext_vector_type(4)));
typedef int          i32x8 __attribute__((ext_vector_type(8)));

#if __has_builtin(__builtin_amdgcn_tensor_load_to_lds)
#define HAVE_TDM 1
#else
#define HAVE_TDM 0
#endif

// ---------------------------------------------------------------- utilities
__global__ void k_cvt_bf16(const float* __restrict__ src, __bf16* __restrict__ dst, int n) {
    for (int i = blockIdx.x * blockDim.x + threadIdx.x; i < n; i += gridDim.x * blockDim.x)
        dst[i] = (__bf16)src[i];
}

__global__ void k_zero_bar(unsigned* bar) {
    if (threadIdx.x == 0) { bar[0] = 0u; bar[1] = 0u; }
}

__global__ void k_copy_h(const float* __restrict__ enc_state, float* __restrict__ h) {
    h[threadIdx.x] = enc_state[threadIdx.x];
}

__global__ void k_embed(const int* __restrict__ cseq, const int* __restrict__ tseq,
                        const int* __restrict__ sos_p, const float* __restrict__ emb,
                        __bf16* __restrict__ xe, __bf16* __restrict__ xd) {
    int t = blockIdx.x;          // SEQ
    int e = threadIdx.x;         // EMB
    int sos = *sos_p;
    int ce = cseq[t];
    int cd = (t == 0) ? sos : tseq[t - 1];
    xe[(size_t)t * EMB + e] = (__bf16)emb[(size_t)ce * EMB + e];
    xd[(size_t)t * EMB + e] = (__bf16)emb[(size_t)cd * EMB + e];
}

__global__ void k_valid(const int* __restrict__ tgt, const int* __restrict__ eos_p,
                        float* __restrict__ valid) {
    if (threadIdx.x == 0 && blockIdx.x == 0) {
        int eos = *eos_p;
        float run = 1.f;
        for (int t = 0; t < SEQ; ++t) { valid[t] = run; if (tgt[t] == eos) run = 0.f; }
    }
}

__global__ void k_states_out(const float* __restrict__ states, const float* __restrict__ valid,
                             float* __restrict__ out) {
    for (int i = blockIdx.x * blockDim.x + threadIdx.x; i < SEQ * HID;
         i += gridDim.x * blockDim.x)
        out[i] = states[i] * valid[i / HID];
}

// ---------------------------------------------------------------- WMMA GEMM
// C[m][n] = (sum_k A[m][k] * B[n][k] + bias[n]) * rowscale[m]
// A: (M,K) bf16 row-major, B: (N,K) bf16 row-major, C: (M,N) f32.
// Block tile 128M x 64N (8 waves as 4Mx2N), wave tile 32x32 (2x2 WMMA tiles).
// B panel is staged into LDS in KC-column chunks by the Tensor Data Mover
// (double buffered, s_wait_tensorcnt); waves consume it with ds_load_b128.
#define KC   128
#define BT_M 128
#define BT_N 64

__device__ __forceinline__ void tdm_issue_b_chunk(const __bf16* g, int K, unsigned lds_off) {
#if HAVE_TDM
    unsigned long long ga = (unsigned long long)(uintptr_t)g;
    u32x4 g0;
    g0[0] = 0x1u;                                              // count=1, user mode
    g0[1] = lds_off;                                           // lds_addr (bytes)
    g0[2] = (unsigned)(ga & 0xFFFFFFFFull);                    // global_addr[31:0]
    g0[3] = (unsigned)((ga >> 32) & 0x1FFFFFFull) | (2u << 30);// addr[56:32] | type=2
    i32x8 g1;
    g1[0] = 1 << 16;                                           // data_size=1 (2 bytes)
    g1[1] = (int)(((unsigned)K & 0xFFFFu) << 16);              // tensor_dim0[15:0]
    g1[2] = (int)((((unsigned)K >> 16) & 0xFFFFu) |            // tensor_dim0[31:16]
                  ((unsigned)BT_N << 16));                     // tensor_dim1[15:0]=64
    g1[3] = (int)((unsigned)KC << 16);                         // tile_dim0 = KC
    g1[4] = BT_N;                                              // tile_dim1=64, tile_dim2=0
    g1[5] = K;                                                 // tensor_dim0_stride[31:0]
    g1[6] = 0;
    g1[7] = 0;
    i32x4 z4 = {0, 0, 0, 0};
#if __clang_major__ >= 23
    i32x8 z8 = {0, 0, 0, 0, 0, 0, 0, 0};
    __builtin_amdgcn_tensor_load_to_lds(g0, g1, z4, z4, z8, 0);
#else
    __builtin_amdgcn_tensor_load_to_lds(g0, g1, z4, z4, 0);
#endif
#else
    (void)g; (void)K; (void)lds_off;
#endif
}

__global__ void __launch_bounds__(256)
k_gemm_bf16(const __bf16* __restrict__ A, const __bf16* __restrict__ B,
            const float* __restrict__ bias, const float* __restrict__ rowscale,
            float* __restrict__ C, int M, int N, int K) {
    __shared__ __align__(16) __bf16 sB[2][BT_N * KC];   // 2 x 16 KB

    const int wave = threadIdx.x >> 5;   // 0..7
    const int lane = threadIdx.x & 31;
    const int wm   = wave >> 1;          // 0..3 : M position in block
    const int wn   = wave & 1;           // 0..1 : N position in block

    const int nBlkN = N / BT_N;
    const int bm = blockIdx.x / nBlkN;
    const int bn = blockIdx.x % nBlkN;

    const int l  = lane & 15;            // A row / B col within tile
    const int hi = lane >> 4;            // K-phase lane group

    const int m0 = bm * BT_M + wm * 32;
    const __bf16* Arow0 = A + (size_t)(m0 + l) * K + hi * 8;        // M-tile 0
    const __bf16* Arow1 = A + (size_t)(m0 + 16 + l) * K + hi * 8;   // M-tile 1
    const __bf16* Bpanel = B + (size_t)(bn * BT_N) * K;

    const unsigned ldsOff[2] = { (unsigned)(uintptr_t)&sB[0][0],
                                 (unsigned)(uintptr_t)&sB[1][0] };

    v8f acc00 = {}, acc01 = {}, acc10 = {}, acc11 = {};
    const int nChunks = K / KC;

#if HAVE_TDM
    if (wave == 0) tdm_issue_b_chunk(Bpanel, K, ldsOff[0]);   // prologue: chunk 0
#endif

    for (int c = 0; c < nChunks; ++c) {
#if HAVE_TDM
        if (wave == 0) __builtin_amdgcn_s_wait_tensorcnt(0);  // chunk c landed
        __syncthreads();
        if (wave == 0 && c + 1 < nChunks)                     // prefetch chunk c+1
            tdm_issue_b_chunk(Bpanel + (size_t)(c + 1) * KC, K, ldsOff[(c + 1) & 1]);
        const __bf16* sbuf = &sB[c & 1][0];
#else
        __syncthreads();
        {   // cooperative fallback copy: 64 x KC bf16 chunk
            for (int i = threadIdx.x; i < BT_N * KC / 8; i += 256) {
                int row = i / (KC / 8), c8 = i % (KC / 8);
                *(uint4*)(&sB[0][row * KC + c8 * 8]) =
                    *(const uint4*)(Bpanel + (size_t)row * K + (size_t)c * KC + c8 * 8);
            }
        }
        __syncthreads();
        const __bf16* sbuf = &sB[0][0];
#endif
        const __bf16* a0 = Arow0 + (size_t)c * KC;
        const __bf16* a1 = Arow1 + (size_t)c * KC;
        const __bf16* b0 = sbuf + (wn * 32 + l) * KC + hi * 8;        // N-tile 0
        const __bf16* b1 = sbuf + (wn * 32 + 16 + l) * KC + hi * 8;   // N-tile 1

        __builtin_prefetch((const void*)(a0 + KC), 0, 1);   // global_prefetch_b8
        __builtin_prefetch((const void*)(a1 + KC), 0, 1);

        #pragma unroll
        for (int kk = 0; kk < KC; kk += 32) {
            v16bf av0, av1, bv0, bv1;
            { uint4* p = (uint4*)&av0; p[0] = *(const uint4*)(a0 + kk);
                                       p[1] = *(const uint4*)(a0 + kk + 16); }
            { uint4* p = (uint4*)&av1; p[0] = *(const uint4*)(a1 + kk);
                                       p[1] = *(const uint4*)(a1 + kk + 16); }
            { uint4* p = (uint4*)&bv0; p[0] = *(const uint4*)(b0 + kk);
                                       p[1] = *(const uint4*)(b0 + kk + 16); }
            { uint4* p = (uint4*)&bv1; p[0] = *(const uint4*)(b1 + kk);
                                       p[1] = *(const uint4*)(b1 + kk + 16); }
            acc00 = __builtin_amdgcn_wmma_f32_16x16x32_bf16(false, av0, false, bv0,
                                                            (short)0, acc00, false, false);
            acc01 = __builtin_amdgcn_wmma_f32_16x16x32_bf16(false, av0, false, bv1,
                                                            (short)0, acc01, false, false);
            acc10 = __builtin_amdgcn_wmma_f32_16x16x32_bf16(false, av1, false, bv0,
                                                            (short)0, acc10, false, false);
            acc11 = __builtin_amdgcn_wmma_f32_16x16x32_bf16(false, av1, false, bv1,
                                                            (short)0, acc11, false, false);
        }
        __syncthreads();   // all waves done with buffer before TDM overwrites it
    }

    // store 2x2 tiles; C layout: vgpr r -> row m0 + hi*8 + r, col = tile col + l
    const int nb = bn * BT_N + wn * 32;
    #pragma unroll
    for (int mt = 0; mt < 2; ++mt) {
        int mb = m0 + mt * 16 + hi * 8;
        #pragma unroll
        for (int nt = 0; nt < 2; ++nt) {
            int n = nb + nt * 16 + l;
            v8f acc = (mt == 0) ? (nt == 0 ? acc00 : acc01)
                                : (nt == 0 ? acc10 : acc11);
            float bs = bias ? bias[n] : 0.f;
            #pragma unroll
            for (int r = 0; r < 8; ++r) {
                float sc = rowscale ? rowscale[mb + r] : 1.f;
                C[(size_t)(mb + r) * N + n] = (acc[r] + bs) * sc;
            }
        }
    }
}

// ---------------------------------------------------------------- recurrence
#define NBLK_R 128
#define TPB_R  384

__device__ __forceinline__ float sigmoidf_(float x) { return 1.f / (1.f + __expf(-x)); }

__global__ void k_gru_seq(const float* __restrict__ gi,       // (SEQ,3H) = x@wih^T+bih
                          const float* __restrict__ whh,      // (3H,H) fp32
                          const float* __restrict__ bhh,      // (3H)
                          const int*   __restrict__ mask_ids, // char_seq or null
                          float*       __restrict__ h,        // (H) live state
                          float*       __restrict__ states,   // (SEQ,H) or null
                          __bf16*      __restrict__ states_bf,// (SEQ,H) or null
                          unsigned*    __restrict__ bar) {
    const int tid      = threadIdx.x;
    const int dotLocal = tid >> 4;        // 0..23  (8 j x 3 gates)
    const int sub      = tid & 15;
    const int jLocal   = dotLocal / 3;
    const int gate     = dotLocal % 3;
    const int j        = blockIdx.x * 8 + jLocal;
    const int row      = gate * HID + j;

    __shared__ __align__(16) float sh[HID];
    __shared__ float sdot[24];

    const float4* w4 = (const float4*)(whh + (size_t)row * HID + sub * 64);
    volatile unsigned* vbar = (volatile unsigned*)bar;

    for (int step = 0; step < SEQ; ++step) {
        for (int i = tid; i < HID; i += TPB_R) sh[i] = h[i];
        __syncthreads();

        float acc = 0.f;
        const float4* h4 = (const float4*)(&sh[sub * 64]);
        #pragma unroll
        for (int k = 0; k < 16; ++k) {
            float4 w = w4[k], x = h4[k];
            acc = fmaf(w.x, x.x, acc); acc = fmaf(w.y, x.y, acc);
            acc = fmaf(w.z, x.z, acc); acc = fmaf(w.w, x.w, acc);
        }
        for (int off = 8; off >= 1; off >>= 1) acc += __shfl_down(acc, off, 16);
        if (sub == 0) sdot[dotLocal] = acc;
        __syncthreads();

        if (tid < 8) {
            int jj = blockIdx.x * 8 + tid;
            float ghr = sdot[tid * 3 + 0] + bhh[jj];
            float ghz = sdot[tid * 3 + 1] + bhh[HID + jj];
            float ghn = sdot[tid * 3 + 2] + bhh[2 * HID + jj];
            const float* g = gi + (size_t)step * G3H;
            float r  = sigmoidf_(g[jj] + ghr);
            float z  = sigmoidf_(g[HID + jj] + ghz);
            float n  = tanhf(g[2 * HID + jj] + r * ghn);
            float ho = sh[jj];
            float hn = (1.f - z) * n + z * ho;
            if (mask_ids && mask_ids[step] == 0) hn = ho;
            h[jj] = hn;
            if (states)    states[(size_t)step * HID + jj]    = hn;
            if (states_bf) states_bf[(size_t)step * HID + jj] = (__bf16)hn;
        }

        __threadfence();
        __syncthreads();
        if (tid == 0) {
            atomicAdd(bar, 1u);
            unsigned target = (unsigned)NBLK_R * (unsigned)(step + 1);
            while (*vbar < target) __builtin_amdgcn_s_sleep(1);
        }
        __syncthreads();
        __threadfence();
    }
}

// ---------------------------------------------------------------- launcher
extern "C" void kernel_launch(void* const* d_in, const int* in_sizes, int n_in,
                              void* d_out, int out_size, void* d_ws, size_t ws_size,
                              hipStream_t stream) {
    const int*   char_seq = (const int*)d_in[0];
    const int*   tgt      = (const int*)d_in[1];
    const float* enc_st   = (const float*)d_in[2];
    const float* emb      = (const float*)d_in[3];
    const float* enc_wih  = (const float*)d_in[4];
    const float* enc_whh  = (const float*)d_in[5];
    const float* enc_bih  = (const float*)d_in[6];
    const float* enc_bhh  = (const float*)d_in[7];
    const float* dec_wih  = (const float*)d_in[8];
    const float* dec_whh  = (const float*)d_in[9];
    const float* dec_bih  = (const float*)d_in[10];
    const float* dec_bhh  = (const float*)d_in[11];
    const float* out_w    = (const float*)d_in[12];
    const float* out_b    = (const float*)d_in[13];
    const int*   sos_p    = (const int*)d_in[14];
    const int*   eos_p    = (const int*)d_in[15];

    uint8_t* ws = (uint8_t*)d_ws;
    size_t off = 0;
    auto alloc = [&](size_t bytes) -> void* {
        void* p = ws + off;
        off += (bytes + 255) & ~(size_t)255;
        return p;
    };
    unsigned* bar        = (unsigned*)alloc(2 * sizeof(unsigned));
    float*    valid      = (float*)   alloc(SEQ * sizeof(float));
    float*    h          = (float*)   alloc(HID * sizeof(float));
    __bf16*   x_enc      = (__bf16*)  alloc((size_t)SEQ * EMB * 2);
    __bf16*   x_dec      = (__bf16*)  alloc((size_t)SEQ * EMB * 2);
    __bf16*   enc_wih_bf = (__bf16*)  alloc((size_t)G3H * EMB * 2);
    __bf16*   dec_wih_bf = (__bf16*)  alloc((size_t)G3H * EMB * 2);
    __bf16*   out_w_bf   = (__bf16*)  alloc((size_t)VOCAB * HID * 2);
    float*    gi_enc     = (float*)   alloc((size_t)SEQ * G3H * 4);
    float*    gi_dec     = (float*)   alloc((size_t)SEQ * G3H * 4);
    float*    states     = (float*)   alloc((size_t)SEQ * HID * 4);
    __bf16*   states_bf  = (__bf16*)  alloc((size_t)SEQ * HID * 2);
    (void)ws_size; (void)n_in; (void)in_sizes; (void)out_size;

    float* out_scores = (float*)d_out;                    // (SEQ, VOCAB)
    float* out_states = out_scores + (size_t)SEQ * VOCAB; // (SEQ, HID)

    k_zero_bar<<<1, 32, 0, stream>>>(bar);

    k_cvt_bf16<<<2048, 256, 0, stream>>>(enc_wih, enc_wih_bf, G3H * EMB);
    k_cvt_bf16<<<2048, 256, 0, stream>>>(dec_wih, dec_wih_bf, G3H * EMB);
    k_cvt_bf16<<<4096, 256, 0, stream>>>(out_w,   out_w_bf,   VOCAB * HID);

    k_embed<<<SEQ, EMB, 0, stream>>>(char_seq, tgt, sos_p, emb, x_enc, x_dec);
    k_copy_h<<<1, HID, 0, stream>>>(enc_st, h);

    {   // Gi = X @ wih^T + bih  (M=1024, N=3072, K=512)
        int blocks = (SEQ / BT_M) * (G3H / BT_N);   // 8 * 48
        k_gemm_bf16<<<blocks, 256, 0, stream>>>(x_enc, enc_wih_bf, enc_bih, nullptr,
                                                gi_enc, SEQ, G3H, EMB);
        k_gemm_bf16<<<blocks, 256, 0, stream>>>(x_dec, dec_wih_bf, dec_bih, nullptr,
                                                gi_dec, SEQ, G3H, EMB);
    }

    k_gru_seq<<<NBLK_R, TPB_R, 0, stream>>>(gi_enc, enc_whh, enc_bhh, char_seq,
                                            h, nullptr, nullptr, bar + 0);
    k_gru_seq<<<NBLK_R, TPB_R, 0, stream>>>(gi_dec, dec_whh, dec_bhh, nullptr,
                                            h, states, states_bf, bar + 1);

    k_valid<<<1, 32, 0, stream>>>(tgt, eos_p, valid);

    {   // scores = (states @ out_w^T + out_b) * valid[t]  (M=1024, N=8000, K=1024)
        int blocks = (SEQ / BT_M) * (VOCAB / BT_N); // 8 * 125
        k_gemm_bf16<<<blocks, 256, 0, stream>>>(states_bf, out_w_bf, out_b, valid,
                                                out_scores, SEQ, VOCAB, HID);
    }

    k_states_out<<<2048, 256, 0, stream>>>(states, valid, out_states);
}
